// Channel_Projection_LoRA_Old_47536698032193
// MI455X (gfx1250) — matrix operand order repaired and checked by reading, not compiled, not run on hardware
//
#include <hip/hip_runtime.h>
#include <hip/hip_bf16.h>

// ---------------------------------------------------------------------------
// Channel projection + LoRA, folded:  out = X @ (U_w + sum_t B_t A_t)^T
// Pass 1: build W_eff (fp32) and split to bf16 hi/lo in d_ws.
// Pass 2: 32768x768x768 GEMM via v_wmma_f32_16x16x32_bf16 with bf16x2
//         error compensation (Ah*Bh + Ah*Bl + Al*Bh).
//         W tiles streamed global->LDS with GLOBAL_LOAD_ASYNC_TO_LDS_B128
//         (ASYNCcnt) when the toolchain exposes the builtin.
// ---------------------------------------------------------------------------

typedef __bf16    v16bf __attribute__((ext_vector_type(16)));
typedef float     v8f   __attribute__((ext_vector_type(8)));
typedef float     v4f   __attribute__((ext_vector_type(4)));
typedef unsigned  v8u   __attribute__((ext_vector_type(8)));
typedef unsigned  v4u   __attribute__((ext_vector_type(4)));

// GCC-vector int4 to match the async-LDS builtin's parameter type exactly
// (diagnostic: "__attribute__((__vector_size__(4 * sizeof(int)))) int __device__ *")
typedef int v4i_gcc __attribute__((vector_size(16)));

#define AS1 __attribute__((address_space(1)))
#define AS3 __attribute__((address_space(3)))

#if __has_builtin(__builtin_amdgcn_global_load_async_to_lds_b128)
#define USE_ASYNC_LDS 1
#else
#define USE_ASYNC_LDS 0
#endif

static constexpr int C   = 768;          // C_in == C_out
static constexpr int MR  = 32 * 32 * 32; // 32768 output rows (B*H*W)
static constexpr int TM  = 128;          // block tile M
static constexpr int TN  = 128;          // block tile N
static constexpr int TK  = 32;           // K step (matches 16x16x32 WMMA)
static constexpr int XS  = TK + 8;       // padded LDS row stride (ushorts) -> bank-conflict free

__device__ __forceinline__ unsigned short f2bf(float f) {
    unsigned u = __float_as_uint(f);
    u += 0x7FFFu + ((u >> 16) & 1u);     // round-to-nearest-even
    return (unsigned short)(u >> 16);
}
__device__ __forceinline__ float bf2f(unsigned short h) {
    return __uint_as_float(((unsigned)h) << 16);
}

// Load a 16-element bf16 fragment as two 16B LDS reads (ds_load_b128 each).
__device__ __forceinline__ v16bf ldfrag(const unsigned short* s, int idx, int off2) {
    v4u a = *(const v4u*)(s + idx);
    v4u b = *(const v4u*)(s + idx + off2);
    v8u c = { a[0], a[1], a[2], a[3], b[0], b[1], b[2], b[3] };
    return __builtin_bit_cast(v16bf, c);
}

__device__ __forceinline__ void wait_asynccnt0() {
#if __has_builtin(__builtin_amdgcn_s_wait_asynccnt)
    __builtin_amdgcn_s_wait_asynccnt(0);
#else
    asm volatile("s_wait_asynccnt 0x0" ::: "memory");
#endif
}

// ---------------------------------------------------------------------------
// Pass 1: W_eff = U_w + sum_{t<=task} B_t @ A_t   (K = 16*(task+1) = 160)
// ---------------------------------------------------------------------------
__global__ void __launch_bounds__(256)
weff_build(const float* __restrict__ Uw,
           const float* __restrict__ A,   // (T,16,C)
           const float* __restrict__ B,   // (T,C,16)
           const int*   __restrict__ taskp,
           unsigned short* __restrict__ Whi,
           unsigned short* __restrict__ Wlo) {
    const int i = blockIdx.x * blockDim.x + threadIdx.x;   // input channel
    const int o = blockIdx.y;                              // output channel
    const int task = *taskp;

    float acc = Uw[(size_t)o * C + i];
    if (task >= 0) {
        for (int t = 0; t <= task; ++t) {
            const float* Bp = B + ((size_t)t * C + o) * 16;  // uniform -> s_loads
            const float* Ap = A + (size_t)t * 16 * C + i;    // coalesced
            #pragma unroll
            for (int r = 0; r < 16; ++r)
                acc = fmaf(Bp[r], Ap[(size_t)r * C], acc);
        }
    } else {
        acc += (o == i) ? 1.0f : 0.0f;                       // task<0: x_lora = x
    }
    const unsigned short hi = f2bf(acc);
    const unsigned short lo = f2bf(acc - bf2f(hi));
    Whi[(size_t)o * C + i] = hi;
    Wlo[(size_t)o * C + i] = lo;
}

// ---------------------------------------------------------------------------
// Pass 2: out[M x C] = X[M x C] @ Weff^T, bf16x2-compensated WMMA GEMM
// Block: 256 threads = 8 waves (wave32). Block tile 128M x 128N.
// Wave grid 4(M) x 2(N); each wave owns a 32M x 64N area = 2x4 accumulators.
// 24 v_wmma per wave per K-step against 12 LDS fragment loads.
// ---------------------------------------------------------------------------
__global__ void __launch_bounds__(256)
lora_gemm(const float* __restrict__ X,
          const unsigned short* __restrict__ Whi,
          const unsigned short* __restrict__ Wlo,
          float* __restrict__ Out) {
    __shared__ __align__(16) unsigned short sXh[TM * XS];
    __shared__ __align__(16) unsigned short sXl[TM * XS];
    __shared__ __align__(16) unsigned short sWh[TN * XS];
    __shared__ __align__(16) unsigned short sWl[TN * XS];
    unsigned* sXh_u = (unsigned*)sXh;
    unsigned* sXl_u = (unsigned*)sXl;
#if !USE_ASYNC_LDS
    unsigned* sWh_u = (unsigned*)sWh;
    unsigned* sWl_u = (unsigned*)sWl;
#endif

    const int tid   = threadIdx.x;
    const int bn    = blockIdx.x;       // 6 N tiles
    const int bm    = blockIdx.y;       // 256 M tiles
    const int lane  = tid & 31;
    const int wave  = tid >> 5;
    const int wm    = (wave & 3) * 32;  // wave M offset within block tile
    const int wn    = (wave >> 2) * 64; // wave N offset within block tile
    const int l16   = lane & 15;
    const bool hiHalf = (lane >= 16);

    v8f acc[2][4] = {};

    for (int k0 = 0; k0 < C; k0 += TK) {
        __syncthreads();

        // ---- stage W tile: 128 x 32 bf16 hi/lo.
        // Data is already bf16 -> no VALU work needed; stream it with the
        // async global->LDS engine (ASYNCcnt) when available.
        #pragma unroll
        for (int i = 0; i < 2; ++i) {
            const int j   = tid + i * 256;     // 16B-chunk index, 512 total
            const int row = j >> 2;            // 4 chunks of 8 ushorts per row
            const int c8  = (j & 3) * 8;       // ushort offset within row
            const size_t g  = (size_t)(bn * TN + row) * C + k0 + c8;
            const int    ub = row * XS + c8;   // ushort offset in LDS
#if USE_ASYNC_LDS
            __builtin_amdgcn_global_load_async_to_lds_b128(
                (AS1 v4i_gcc*)(Whi + g), (AS3 v4i_gcc*)&sWh[ub], 0, 0);
            __builtin_amdgcn_global_load_async_to_lds_b128(
                (AS1 v4i_gcc*)(Wlo + g), (AS3 v4i_gcc*)&sWl[ub], 0, 0);
#else
            const v4u wh = *(const v4u*)(Whi + g);
            const v4u wl = *(const v4u*)(Wlo + g);
            const int uu = (ub >> 1);
            sWh_u[uu] = wh[0]; sWh_u[uu + 1] = wh[1];
            sWh_u[uu + 2] = wh[2]; sWh_u[uu + 3] = wh[3];
            sWl_u[uu] = wl[0]; sWl_u[uu + 1] = wl[1];
            sWl_u[uu + 2] = wl[2]; sWl_u[uu + 3] = wl[3];
#endif
        }

        // ---- stage X tile: 128 x 32 fp32 -> bf16 hi/lo (4 float4 per thread)
        #pragma unroll
        for (int i = 0; i < 4; ++i) {
            const int j   = tid + i * 256;     // float4 index, 1024 total
            const int row = j >> 3;            // 8 float4 per row
            const int kc  = (j & 7) * 4;
            const v4f v = *(const v4f*)(X + (size_t)(bm * TM + row) * C + k0 + kc);
            unsigned short h[4], l[4];
            #pragma unroll
            for (int e = 0; e < 4; ++e) {
                h[e] = f2bf(v[e]);
                l[e] = f2bf(v[e] - bf2f(h[e]));
            }
            const int ub = row * (XS / 2) + (kc >> 1);
            sXh_u[ub]     = (unsigned)h[0] | ((unsigned)h[1] << 16);
            sXh_u[ub + 1] = (unsigned)h[2] | ((unsigned)h[3] << 16);
            sXl_u[ub]     = (unsigned)l[0] | ((unsigned)l[1] << 16);
            sXl_u[ub + 1] = (unsigned)l[2] | ((unsigned)l[3] << 16);
        }

        // Near-cache prefetch of next K slab of X (global_prefetch_b8)
        __builtin_prefetch(X + (size_t)(bm * TM + (tid >> 1)) * C + k0 + TK + (tid & 1) * 16, 0, 3);

#if USE_ASYNC_LDS
        wait_asynccnt0();                  // my async W chunks have landed
#endif
        __syncthreads();                   // everyone's LDS data visible

        // ---- build fragments (ISA 16-bit A layout: lanes<16 K{0-7,16-23})
        const int kbA = hiHalf ? 8 : 0;
        const int kbB = hiHalf ? 8 : 0;
        v16bf axh[2], axl[2], bwh[4], bwl[4];
        #pragma unroll
        for (int mt = 0; mt < 2; ++mt) {
            const int r = (wm + mt * 16 + l16) * XS + kbA;
            axh[mt] = ldfrag(sXh, r, 16);
            axl[mt] = ldfrag(sXl, r, 16);
        }
        #pragma unroll
        for (int nt = 0; nt < 4; ++nt) {
            const int r = (wn + nt * 16 + l16) * XS + kbB;
            bwh[nt] = ldfrag(sWh, r, 16);
            bwl[nt] = ldfrag(sWl, r, 16);
        }

        // ---- 24 WMMAs: Ah*Bh + Ah*Bl + Al*Bh (lo*lo dropped)
        #pragma unroll
        for (int mt = 0; mt < 2; ++mt) {
            #pragma unroll
            for (int nt = 0; nt < 4; ++nt) {
                acc[mt][nt] = __builtin_amdgcn_wmma_f32_16x16x32_bf16(
                    false, axh[mt], false, bwh[nt], (short)0, acc[mt][nt], false, false);
                acc[mt][nt] = __builtin_amdgcn_wmma_f32_16x16x32_bf16(
                    false, axh[mt], false, bwl[nt], (short)0, acc[mt][nt], false, false);
                acc[mt][nt] = __builtin_amdgcn_wmma_f32_16x16x32_bf16(
                    false, axl[mt], false, bwh[nt], (short)0, acc[mt][nt], false, false);
            }
        }
    }

    // ---- epilogue: C/D layout => VGPR v holds row v (+8 for lanes>=16), col = lane%16
    #pragma unroll
    for (int mt = 0; mt < 2; ++mt) {
        const int mrow0 = bm * TM + wm + mt * 16 + (hiHalf ? 8 : 0);
        #pragma unroll
        for (int nt = 0; nt < 4; ++nt) {
            const int ncol = bn * TN + wn + nt * 16 + l16;
            #pragma unroll
            for (int v = 0; v < 8; ++v)
                Out[(size_t)(mrow0 + v) * C + ncol] = acc[mt][nt][v];
        }
    }
}

// ---------------------------------------------------------------------------
extern "C" void kernel_launch(void* const* d_in, const int* in_sizes, int n_in,
                              void* d_out, int out_size, void* d_ws, size_t ws_size,
                              hipStream_t stream) {
    const float* x      = (const float*)d_in[0];   // (32,32,32,768)
    const float* U_w    = (const float*)d_in[1];   // (768,768)
    const float* lora_A = (const float*)d_in[2];   // (10,16,768)
    const float* lora_B = (const float*)d_in[3];   // (10,768,16)
    const int*   task   = (const int*)d_in[4];     // scalar
    float* out = (float*)d_out;

    unsigned short* Whi = (unsigned short*)d_ws;          // 768*768 bf16 hi
    unsigned short* Wlo = Whi + (size_t)C * C;            // 768*768 bf16 lo (residual)

    // Pass 1: fold LoRA into the base weight, split to bf16 hi/lo.
    dim3 g1(C / 256, C, 1);
    weff_build<<<g1, 256, 0, stream>>>(U_w, lora_A, lora_B, task, Whi, Wlo);

    // Pass 2: the big WMMA GEMM.
    dim3 g2(C / TN, MR / TM, 1);                          // (6, 256)
    lora_gemm<<<g2, 256, 0, stream>>>(x, Whi, Wlo, out);
}
